// Attention_46110768890511
// MI455X (gfx1250) — matrix operand order repaired and checked by reading, not compiled, not run on hardware
//
#include <hip/hip_runtime.h>

typedef __attribute__((ext_vector_type(16))) _Float16 v16h;
typedef __attribute__((ext_vector_type(8)))  float    v8f;

__device__ __forceinline__ v8f wmma16(v16h a, v16h b, v8f c) {
  return __builtin_amdgcn_wmma_f32_16x16x32_f16(false, a, false, b, (short)0, c, false, false);
}

// ---- geometry ----
// B=8, C=192, H=W=128, HW=16384, HEADS=4, DH=48
// GEMM1/3: M=192 (12 mt), K=192 (6 kt of 32), N=16384 (1024 nt of 16)
// attn S:  M=N=48 (3 tiles), K=16384 (512 kt)
// out:     M=48, K=64 (48 padded), N=16384
// fragment = 512 f16 (32 lanes x 16), 1KB

// -------- zero helper --------
__global__ void k_zero(uint4* p, long n) {
  long i = (long)blockIdx.x * blockDim.x + threadIdx.x;
  if (i < n) p[i] = make_uint4(0u, 0u, 0u, 0u);
}

// -------- pack 4 weight matrices [192x192] into A-fragments --------
__global__ void k_pack_w(const float* wq, const float* wk, const float* wv,
                         const float* wp, _Float16* WA) {
  int idx = blockIdx.x * 256 + threadIdx.x;        // 4*12*6*512 = 147456
  int wid = idx / 36864;
  int r   = idx % 36864;
  int frag = r >> 9;                                // mt*6+kt
  int lane = (r >> 4) & 31;
  int e    = r & 15;
  int mt = frag / 6, kt = frag % 6;
  int m = mt * 16 + (lane & 15);
  int k = e + 16 * (lane >> 4);  // A-frag K order
  int c = kt * 32 + k;
  const float* src = (wid == 0) ? wq : (wid == 1) ? wk : (wid == 2) ? wv : wp;
  WA[idx] = (_Float16)src[m * 192 + c];
}

// -------- LayerNorm over channels, pack B-fragments (f16) --------
__global__ void k_ln(const float* xx, const float* qin, const float* lw,
                     const float* lb, _Float16* XP, _Float16* QP) {
  int t  = blockIdx.x >> 9;                  // 0: xx->XP, 1: q_in->QP
  int pg = blockIdx.x & 511;
  int p  = pg * 256 + threadIdx.x;           // global pixel 0..131071
  int b  = p >> 14;
  int n  = p & 16383;
  const float* src = t ? qin : xx;
  _Float16*    dst = t ? QP : XP;
  const float* base = src + ((long)b * 192) * 16384 + n;
  float s = 0.f, ss = 0.f;
  for (int c = 0; c < 192; ++c) { float v = base[(long)c * 16384]; s += v; ss += v * v; }
  float mu   = s * (1.0f / 192.0f);
  float var  = ss * (1.0f / 192.0f) - mu * mu;
  float rstd = rsqrtf(var + 1e-5f);
  long fb = ((long)b * 1024 + (n >> 4)) * 6;
  int  nl = n & 15;
  for (int c = 0; c < 192; ++c) {
    float y = (base[(long)c * 16384] - mu) * rstd * lw[c] + lb[c];
    int kt = c >> 5, half = (c >> 4) & 1, e = c & 15;   // B-frag K order
    dst[(fb + kt) * 512 + (nl + 16 * half) * 16 + e] = (_Float16)y;
  }
}

// -------- QKV projection GEMMs: one wave -> 16x64 tile --------
__global__ void k_gemm1(const _Float16* WA, const _Float16* XP, const _Float16* QP,
                        _Float16* QC, _Float16* KC, _Float16* VC) {
  int wave = blockIdx.x * (blockDim.x >> 5) + (threadIdx.x >> 5);
  int lane = threadIdx.x & 31;
  int ng = wave & 255; int tmp = wave >> 8;
  int mt = tmp % 12; tmp /= 12;
  int b = tmp & 7; int t = tmp >> 3;                 // t: 0=q,1=k,2=v
  const _Float16* wsrc = WA + t * 36864;
  const _Float16* xsrc = (t == 0) ? QP : XP;
  _Float16* dst = (t == 0) ? QC : (t == 1) ? KC : VC;
  int nt0 = ng * 4;
  v8f acc[4] = {};
  for (int kt = 0; kt < 6; ++kt) {
    v16h A = *(const v16h*)(wsrc + ((mt * 6 + kt) * 512 + lane * 16));
    for (int j = 0; j < 4; ++j) {
      v16h Bf = *(const v16h*)(xsrc + ((((long)b * 1024 + nt0 + j) * 6 + kt) * 512 + lane * 16));
      acc[j] = wmma16(A, Bf, acc[j]);
    }
  }
  int half = lane >> 4, nl = lane & 15;
  for (int j = 0; j < 4; ++j)
    for (int r = 0; r < 8; ++r) {
      int m = mt * 16 + r + 8 * half;
      long n = (long)(nt0 + j) * 16 + nl;
      dst[((long)b * 192 + m) * 16384 + n] = (_Float16)acc[j][r];
    }
}

// -------- depthwise 3x3 + pack q/k/v for attention + sum-of-squares --------
__global__ void k_dw(const _Float16* QC, const _Float16* KC, const _Float16* VC,
                     const float* wdq, const float* wdk, const float* wdv,
                     _Float16* QA, _Float16* KB, _Float16* VB,
                     float* nq2, float* nk2) {
  int pg = blockIdx.x & 63;
  int tmp = blockIdx.x >> 6;
  int c = tmp % 192; tmp /= 192;
  int b = tmp & 7; int t = tmp >> 3;
  int pix = pg * 256 + threadIdx.x;
  int y = pix >> 7, x = pix & 127;
  const _Float16* in = (t == 0) ? QC : (t == 1) ? KC : VC;
  const float*    wd = (t == 0) ? wdq : (t == 1) ? wdk : wdv;
  const _Float16* ib = in + ((long)b * 192 + c) * 16384;
  float acc = 0.f;
  for (int dy = -1; dy <= 1; ++dy) {
    int yy = y + dy; if (yy < 0 || yy > 127) continue;
    for (int dx = -1; dx <= 1; ++dx) {
      int xc = x + dx; if (xc < 0 || xc > 127) continue;
      acc += wd[c * 9 + (dy + 1) * 3 + (dx + 1)] * (float)ib[yy * 128 + xc];
    }
  }
  int head = c / 48, cl = c % 48;
  long bh = b * 4 + head;
  int kt = pix >> 5, kl = pix & 31;
  if (t == 0) {                                   // q -> A-fragments (K = hw)
    int half = kl >> 4;
    int e = kl & 15;
    int lo = (cl & 15) + 16 * half;
    QA[bh * 786432 + ((long)kt * 3 + (cl >> 4)) * 512 + lo * 16 + e] = (_Float16)acc;
  } else if (t == 1) {                            // k -> B-fragments (K = hw)
    int lo = (cl & 15) + 16 * (kl >> 4);
    KB[bh * 786432 + ((long)kt * 3 + (cl >> 4)) * 512 + lo * 16 + (kl & 15)] = (_Float16)acc;
  } else {                                        // v -> B-fragments (K = ch, pad 64)
    int ck = cl & 31, ckt = cl >> 5;
    int lo = (pix & 15) + 16 * (ck >> 4);
    VB[bh * 1048576 + ((long)(pix >> 4) * 2 + ckt) * 512 + lo * 16 + (ck & 15)] = (_Float16)acc;
  }
  __shared__ float red[256];
  if (t < 2) {
    red[threadIdx.x] = acc * acc;
    __syncthreads();
    for (int s = 128; s > 0; s >>= 1) {
      if (threadIdx.x < s) red[threadIdx.x] += red[threadIdx.x + s];
      __syncthreads();
    }
    if (threadIdx.x == 0) atomicAdd((t == 0 ? nq2 : nk2) + b * 192 + c, red[0]);
  }
}

// -------- S = q . k^T  (48x48, K split over 32 chunks, f32 atomics) --------
__global__ void k_gemmS(const _Float16* QA, const _Float16* KB, float* S) {
  int bh = blockIdx.x >> 5;
  int kc = blockIdx.x & 31;
  int lane = threadIdx.x;
  const _Float16* qa = QA + (long)bh * 786432;
  const _Float16* kb = KB + (long)bh * 786432;
  v8f acc[3][3] = {};
  for (int kt = kc * 16; kt < kc * 16 + 16; ++kt) {
    v16h A[3], Bf[3];
    for (int i = 0; i < 3; ++i) A[i]  = *(const v16h*)(qa + ((long)kt * 3 + i) * 512 + lane * 16);
    for (int i = 0; i < 3; ++i) Bf[i] = *(const v16h*)(kb + ((long)kt * 3 + i) * 512 + lane * 16);
    for (int mi = 0; mi < 3; ++mi)
      for (int ni = 0; ni < 3; ++ni)
        acc[mi][ni] = wmma16(A[mi], Bf[ni], acc[mi][ni]);
  }
  int half = lane >> 4, nl = lane & 15;
  float* sb = S + (long)bh * 2304;
  for (int mi = 0; mi < 3; ++mi)
    for (int ni = 0; ni < 3; ++ni)
      for (int r = 0; r < 8; ++r)
        atomicAdd(sb + (mi * 16 + r + 8 * half) * 48 + ni * 16 + nl, acc[mi][ni][r]);
}

// -------- scale by temp/(|q||k|), softmax, emit P as packed A-frags --------
__global__ void k_softmax(const float* S, const float* nq2, const float* nk2,
                          const float* temp, _Float16* P) {
  int bh = blockIdx.x;
  int b = bh >> 2, head = bh & 3;
  int row = threadIdx.x;
  if (row >= 48) return;
  const float* sb  = S + (long)bh * 2304 + row * 48;
  const float* nqb = nq2 + b * 192 + head * 48;
  const float* nkb = nk2 + b * 192 + head * 48;
  float rq = 1.0f / fmaxf(sqrtf(nqb[row]), 1e-12f);
  float tp = temp[head];
  float v[48];
  float mx = -3.4e38f;
  for (int j = 0; j < 48; ++j) {
    float rk = 1.0f / fmaxf(sqrtf(nkb[j]), 1e-12f);
    v[j] = sb[j] * rq * rk * tp;
    mx = fmaxf(mx, v[j]);
  }
  float sum = 0.f;
  for (int j = 0; j < 48; ++j) { v[j] = __expf(v[j] - mx); sum += v[j]; }
  float inv = 1.0f / sum;
  _Float16* pb = P + (long)bh * 3072;
  int mt = row >> 4, lr = row & 15;
  for (int j = 0; j < 64; ++j) {                 // pad K 48..63 with zeros
    float pv = (j < 48) ? v[j] * inv : 0.f;
    int kt = j >> 5, kl = j & 31;
    int half = kl >> 4;
    int e = kl & 15;
    pb[((long)kt * 3 + mt) * 512 + (lr + 16 * half) * 16 + e] = (_Float16)pv;
  }
}

// -------- out = P . v, written as B-frags for the projection GEMM --------
__global__ void k_gemm2(const _Float16* P, const _Float16* VB, _Float16* OP) {
  int wave = blockIdx.x * (blockDim.x >> 5) + (threadIdx.x >> 5);
  int lane = threadIdx.x & 31;
  int ng = wave & 255;
  int bh = wave >> 8;
  int b = bh >> 2, head = bh & 3;
  const _Float16* pb = P + (long)bh * 3072;
  const _Float16* vb = VB + (long)bh * 1048576;
  v16h A[3][2];
  for (int mt = 0; mt < 3; ++mt)
    for (int kt = 0; kt < 2; ++kt)
      A[mt][kt] = *(const v16h*)(pb + ((long)kt * 3 + mt) * 512 + lane * 16);
  int nt0 = ng * 4;
  v8f acc[3][4] = {};
  for (int kt = 0; kt < 2; ++kt)
    for (int j = 0; j < 4; ++j) {
      v16h Bf = *(const v16h*)(vb + ((long)(nt0 + j) * 2 + kt) * 512 + lane * 16);
      for (int mt = 0; mt < 3; ++mt)
        acc[mt][j] = wmma16(A[mt][kt], Bf, acc[mt][j]);
    }
  int half = lane >> 4, nl = lane & 15;
  for (int mt = 0; mt < 3; ++mt)
    for (int j = 0; j < 4; ++j)
      for (int r = 0; r < 8; ++r) {
        int ch = head * 48 + mt * 16 + r + 8 * half;
        OP[(((long)b * 1024 + (nt0 + j)) * 6 + (ch >> 5)) * 512 +
           (nl + 16 * ((ch >> 4) & 1)) * 16 + (ch & 15)] = (_Float16)acc[mt][j][r];
      }
}

// -------- projection GEMM + residual -> f32 output --------
__global__ void k_gemm3(const _Float16* WA, const _Float16* OP, const float* qin,
                        float* out) {
  int wave = blockIdx.x * (blockDim.x >> 5) + (threadIdx.x >> 5);
  int lane = threadIdx.x & 31;
  int ng = wave & 255; int tmp = wave >> 8;
  int mt = tmp % 12; int b = tmp / 12;
  const _Float16* wsrc = WA + 3 * 36864;
  int nt0 = ng * 4;
  v8f acc[4] = {};
  for (int kt = 0; kt < 6; ++kt) {
    v16h A = *(const v16h*)(wsrc + ((mt * 6 + kt) * 512 + lane * 16));
    for (int j = 0; j < 4; ++j) {
      v16h Bf = *(const v16h*)(OP + ((((long)b * 1024 + nt0 + j) * 6 + kt) * 512 + lane * 16));
      acc[j] = wmma16(A, Bf, acc[j]);
    }
  }
  int half = lane >> 4, nl = lane & 15;
  for (int j = 0; j < 4; ++j)
    for (int r = 0; r < 8; ++r) {
      int m = mt * 16 + r + 8 * half;
      long idx = ((long)b * 192 + m) * 16384 + (long)(nt0 + j) * 16 + nl;
      out[idx] = acc[j][r] + qin[idx];
    }
}

extern "C" void kernel_launch(void* const* d_in, const int* in_sizes, int n_in,
                              void* d_out, int out_size, void* d_ws, size_t ws_size,
                              hipStream_t stream) {
  (void)in_sizes; (void)n_in; (void)out_size; (void)ws_size;
  const float* xx     = (const float*)d_in[0];
  const float* q_in   = (const float*)d_in[1];
  const float* ln_w   = (const float*)d_in[2];
  const float* ln_b   = (const float*)d_in[3];
  const float* w_q    = (const float*)d_in[4];
  const float* w_k    = (const float*)d_in[5];
  const float* w_v    = (const float*)d_in[6];
  const float* wd_q   = (const float*)d_in[7];
  const float* wd_k   = (const float*)d_in[8];
  const float* wd_v   = (const float*)d_in[9];
  const float* w_proj = (const float*)d_in[10];
  const float* temp   = (const float*)d_in[11];
  float* out = (float*)d_out;
  char* ws = (char*)d_ws;

  const unsigned long long PACK1 = 50331648ull;          // one packed [192 x 16384] f16 set (per 8 batches)
  _Float16* XP = (_Float16*)(ws);
  _Float16* QP = (_Float16*)(ws + PACK1);
  _Float16* QC = (_Float16*)(ws + 2 * PACK1);
  _Float16* KC = (_Float16*)(ws + 3 * PACK1);
  _Float16* VC = (_Float16*)(ws + 4 * PACK1);
  _Float16* VB = (_Float16*)(ws + 5 * PACK1);            // 67108864 B (K padded to 64)
  _Float16* WA = (_Float16*)(ws + 5 * PACK1 + 67108864ull);     // 294912 B
  float*    S  = (float*)   (ws + 5 * PACK1 + 67108864ull + 294912ull);
  float*    NQ = (float*)((char*)S + 294912);
  float*    NK = NQ + 1536;
  _Float16* P  = (_Float16*)((char*)NK + 6144);
  _Float16* QA = XP;   // reuse: XP dead after k_gemm1
  _Float16* KB = QP;   // reuse: QP dead after k_gemm1
  _Float16* OP = QC;   // reuse: QC dead after k_dw

  k_zero   <<<16384, 256, 0, stream>>>((uint4*)VB, 4194304L);   // VB padding
  k_zero   <<<75,    256, 0, stream>>>((uint4*)S, 19200L);      // S + NQ + NK
  k_pack_w <<<576,   256, 0, stream>>>(w_q, w_k, w_v, w_proj, WA);
  k_ln     <<<1024,  256, 0, stream>>>(xx, q_in, ln_w, ln_b, XP, QP);
  k_gemm1  <<<18432, 128, 0, stream>>>(WA, XP, QP, QC, KC, VC);
  k_dw     <<<294912,256, 0, stream>>>(QC, KC, VC, wd_q, wd_k, wd_v, QA, KB, VB, NQ, NK);
  k_gemmS  <<<1024,  32,  0, stream>>>(QA, KB, S);
  k_softmax<<<32,    64,  0, stream>>>(S, NQ, NK, temp, P);
  k_gemm2  <<<2048,  128, 0, stream>>>(P, VB, OP);
  k_gemm3  <<<6144,  128, 0, stream>>>(WA, OP, q_in, out);
}
